// DyGraphConv2d_69509750718745
// MI455X (gfx1250) — matrix operand order repaired and checked by reading, not compiled, not run on hardware
//
#include <hip/hip_runtime.h>

typedef __bf16 bf16;
typedef __attribute__((ext_vector_type(16))) bf16  v16bf;
typedef __attribute__((ext_vector_type(8)))  float v8f;

#define B_   8
#define C_   192
#define N_   3136
#define K_   9
#define OC_  384
#define NT_  (N_ / 16)   // 196 column/point tiles

union BV { v16bf v; uint4 u[2]; bf16 e[16]; };

__device__ __forceinline__ v8f wmma_bf16(v16bf a, v16bf b, v8f c) {
  // D = A(16x32 bf16) * B(32x16 bf16) + C(16x16 f32)
  return __builtin_amdgcn_wmma_f32_16x16x32_bf16(false, a, false, b, (short)0, c,
                                                 false, false);
}

// ---------------------------------------------------------------------------
// Stage 1: per-point norm; write bf16 point-major copies (normalized + raw)
// and f32 sum-of-squares of the normalized vector (for exact dist ranking).
// ---------------------------------------------------------------------------
__global__ void prep_kernel(const float* __restrict__ x, bf16* __restrict__ xn,
                            bf16* __restrict__ xt, float* __restrict__ sq) {
  int i = blockIdx.x * blockDim.x + threadIdx.x;  // i = b*N + n
  if (i >= B_ * N_) return;
  int b = i / N_, n = i % N_;
  const float* xp = x + (size_t)b * C_ * N_ + n;
  float ss = 0.f;
  for (int c = 0; c < C_; ++c) { float v = xp[(size_t)c * N_]; ss += v * v; }
  float inv = 1.0f / fmaxf(sqrtf(ss), 1e-12f);
  bf16* xnp = xn + (size_t)i * C_;
  bf16* xtp = xt + (size_t)i * C_;
  float s2 = 0.f;
  for (int c = 0; c < C_; ++c) {
    float v  = xp[(size_t)c * N_];
    float vn = v * inv;
    s2 += vn * vn;
    xnp[c] = (bf16)vn;
    xtp[c] = (bf16)v;
  }
  sq[i] = s2;
}

// ---------------------------------------------------------------------------
// Stage 2: fused Gram-matrix + top-9. One wave per 16-row block.
// Ranking key per row n: dot(xn_n, xn_m) - 0.5*sq_m  (== -dist/2 + const).
// Double-buffered B fragments: loads for tile ct+1 issued before the top-k
// scan of tile ct, so load latency hides under the scan.
// ---------------------------------------------------------------------------
__global__ __launch_bounds__(32) void knn_kernel(const bf16* __restrict__ xn,
                                                 const float* __restrict__ sq,
                                                 int* __restrict__ nn) {
  __shared__ float tile[16][17];
  __shared__ float sqs[16];
  const int b = blockIdx.y;
  const int rowBase = blockIdx.x * 16;
  const int lane = threadIdx.x;
  const int lr = lane & 15;
  const bool hi = lane >= 16;
  const int r0 = hi ? 8 : 0;
  const bf16* xb = xn + (size_t)b * N_ * C_;

  // Resident A fragments: rows rowBase..rowBase+15, all 192 channels.
  // A fragments here use contiguous k: lanes 0-15 hold K 0-15, lanes 16-31 K 16-31.
  const bf16* rp = xb + (size_t)(rowBase + lr) * C_ + (hi ? 16 : 0);
  BV A[6];
#pragma unroll
  for (int t = 0; t < 6; ++t) {
    A[t].u[0] = *(const uint4*)(rp + 32 * t);
    A[t].u[1] = *(const uint4*)(rp + 32 * t + 8);
  }

  float val[9]; int idx9[9];
#pragma unroll
  for (int q = 0; q < 9; ++q) { val[q] = -3.4e38f; idx9[q] = 0; }

  BV Bb0[6], Bb1[6];
  float sv0 = 0.f, sv1 = 0.f;

  auto loadB = [&](BV (&Bt)[6], float& sv, int ct) {
    // B fragment: lanes 0-15 hold K 0-15 of column ct*16+lr, lanes 16-31 K 16-31.
    const bf16* cp = xb + (size_t)(ct * 16 + lr) * C_ + (hi ? 16 : 0);
#pragma unroll
    for (int t = 0; t < 6; ++t) {
      Bt[t].u[0] = *(const uint4*)(cp + 32 * t);
      Bt[t].u[1] = *(const uint4*)(cp + 32 * t + 8);
    }
    sv = sq[(size_t)b * N_ + ct * 16 + lr];
  };

  auto step = [&](BV (&Bt)[6], float sv, BV (&Bn)[6], float& svn, int ct) {
    v8f c = {0.f, 0.f, 0.f, 0.f, 0.f, 0.f, 0.f, 0.f};
#pragma unroll
    for (int t = 0; t < 6; ++t) c = wmma_bf16(A[t].v, Bt[t].v, c);
    __syncthreads();  // previous tile fully consumed
#pragma unroll
    for (int r = 0; r < 8; ++r) tile[r0 + r][lr] = c[r];
    if (!hi) sqs[lr] = sv;
    __syncthreads();
    if (ct + 1 < NT_) loadB(Bn, svn, ct + 1);  // hide under the scan below
    if (!hi) {  // lanes 0-15 each own one row; scan 16 candidates
      const int colBase = ct * 16;
      for (int j = 0; j < 16; ++j) {
        float s = tile[lr][j] - 0.5f * sqs[j];
        int cidx = colBase + j;
        if (s > val[8]) {  // rare after warmup
#pragma unroll
          for (int p = 8; p > 0; --p) {
            bool gp = s > val[p - 1];
            float nv = gp ? val[p - 1] : s;
            int   ni = gp ? idx9[p - 1] : cidx;
            bool gc = s > val[p];
            val[p]  = gc ? nv : val[p];
            idx9[p] = gc ? ni : idx9[p];
          }
          if (s > val[0]) { val[0] = s; idx9[0] = cidx; }
        }
      }
    }
  };

  loadB(Bb0, sv0, 0);
  for (int ct = 0; ct < NT_; ct += 2) {
    step(Bb0, sv0, Bb1, sv1, ct);
    step(Bb1, sv1, Bb0, sv0, ct + 1);
  }

  if (!hi) {
    int* op = nn + ((size_t)b * N_ + rowBase + lr) * K_;
#pragma unroll
    for (int q = 0; q < 9; ++q) op[q] = idx9[q];
  }
}

// ---------------------------------------------------------------------------
// Stage 3: grouped 1x1 conv + ReLU + mean-over-K.
// 4 waves per block share one LDS weight staging; each wave owns 16 points.
// grid.z = group. g<2: x_i path (k-independent). g>=2: (x_j - x_i) path,
// with neighbor gathers software-pipelined one k ahead and LDS weight
// fragments staged one M-tile ahead.
// ---------------------------------------------------------------------------
__global__ __launch_bounds__(128) void conv_kernel(const bf16* __restrict__ xt,
                                                   const float* __restrict__ W,
                                                   const float* __restrict__ bias,
                                                   const int* __restrict__ nn,
                                                   float* __restrict__ out) {
  __shared__ bf16 Wl[96 * 96];
  __shared__ float bl[96];
  const int g = blockIdx.z;
  const int b = blockIdx.y;
  const int tid = threadIdx.x;
  const int wave = tid >> 5;
  const int lane = tid & 31;
  const int lr = lane & 15;
  const bool hi = lane >= 16;
  const int r0 = hi ? 8 : 0;
  const int nBase = blockIdx.x * 64 + wave * 16;

  const float* Wg = W + (size_t)g * 96 * 96;  // rows g*96.., 96 cols each
  for (int i = tid; i < 96 * 96; i += 128) Wl[i] = (bf16)Wg[i];
  for (int i = tid; i < 96; i += 128) bl[i] = bias[g * 96 + i];
  __syncthreads();

  const int cbase = (g & 1) * 96;  // input-channel offset inside C=192
  const bf16* xb = xt + (size_t)b * N_ * C_;
  const int col = nBase + lr;
  const bf16* np = xb + (size_t)col * C_ + cbase + (hi ? 16 : 0);

  BV an[3];
#pragma unroll
  for (int t = 0; t < 3; ++t) {
    an[t].u[0] = *(const uint4*)(np + 32 * t);
    an[t].u[1] = *(const uint4*)(np + 32 * t + 8);
  }

  const v8f vzero = {0.f, 0.f, 0.f, 0.f, 0.f, 0.f, 0.f, 0.f};

  auto loadA = [&](BV (&A3)[3], int m) {
#pragma unroll
    for (int t = 0; t < 3; ++t) {
      const bf16* wp = &Wl[(16 * m + lr) * 96 + 32 * t + (hi ? 16 : 0)];
      A3[t].u[0] = *(const uint4*)wp;
      A3[t].u[1] = *(const uint4*)(wp + 8);
    }
  };

  if (g < 2) {  // center path: out[oc] = relu(W*x_i + b), mean is identity
    BV Aa[3], Ab[3];
    loadA(Aa, 0);
#pragma unroll
    for (int m = 0; m < 6; m += 2) {
      loadA(Ab, m + 1);
      v8f c = vzero;
#pragma unroll
      for (int t = 0; t < 3; ++t) c = wmma_bf16(Aa[t].v, an[t].v, c);
      float* op = out + ((size_t)b * OC_ + g * 96 + 16 * m + r0) * N_ + col;
#pragma unroll
      for (int r = 0; r < 8; ++r) {
        float v = c[r] + bl[16 * m + r0 + r];
        op[(size_t)r * N_] = v > 0.f ? v : 0.f;
      }
      if (m + 2 < 6) loadA(Aa, m + 2);
      c = vzero;
#pragma unroll
      for (int t = 0; t < 3; ++t) c = wmma_bf16(Ab[t].v, an[t].v, c);
      op = out + ((size_t)b * OC_ + g * 96 + 16 * (m + 1) + r0) * N_ + col;
#pragma unroll
      for (int r = 0; r < 8; ++r) {
        float v = c[r] + bl[16 * (m + 1) + r0 + r];
        op[(size_t)r * N_] = v > 0.f ? v : 0.f;
      }
    }
  } else {  // diff path: mean_k relu(W*(x_j - x_i) + b)
    v8f acc[6];
#pragma unroll
    for (int m = 0; m < 6; ++m) acc[m] = vzero;

    const int* ip = nn + ((size_t)b * N_ + col) * K_;
    int jx[9];
#pragma unroll
    for (int k = 0; k < 9; ++k) jx[k] = ip[k];

    BV aj0[3], aj1[3];
    auto gather = [&](BV (&aj)[3], int j) {
      const bf16* jp = xb + (size_t)j * C_ + cbase + (hi ? 16 : 0);
#pragma unroll
      for (int t = 0; t < 3; ++t) {
        aj[t].u[0] = *(const uint4*)(jp + 32 * t);
        aj[t].u[1] = *(const uint4*)(jp + 32 * t + 8);
      }
    };

    auto kstep = [&](BV (&aj)[3], BV (&ajn)[3], int k) {
      if (k + 1 < 9) gather(ajn, jx[k + 1]);  // prefetch next neighbor rows
      BV Bt[3];
#pragma unroll
      for (int t = 0; t < 3; ++t)
#pragma unroll
        for (int e = 0; e < 16; ++e)
          Bt[t].e[e] = (bf16)((float)aj[t].e[e] - (float)an[t].e[e]);

      BV Aa[3], Ab[3];
      loadA(Aa, 0);
#pragma unroll
      for (int m = 0; m < 6; m += 2) {
        loadA(Ab, m + 1);
        v8f c = vzero;
#pragma unroll
        for (int t = 0; t < 3; ++t) c = wmma_bf16(Aa[t].v, Bt[t].v, c);
#pragma unroll
        for (int r = 0; r < 8; ++r) {
          float v = c[r] + bl[16 * m + r0 + r];
          acc[m][r] += (v > 0.f ? v : 0.f);
        }
        if (m + 2 < 6) loadA(Aa, m + 2);
        c = vzero;
#pragma unroll
        for (int t = 0; t < 3; ++t) c = wmma_bf16(Ab[t].v, Bt[t].v, c);
#pragma unroll
        for (int r = 0; r < 8; ++r) {
          float v = c[r] + bl[16 * (m + 1) + r0 + r];
          acc[m + 1][r] += (v > 0.f ? v : 0.f);
        }
      }
    };

    gather(aj0, jx[0]);
#pragma unroll
    for (int kk = 0; kk < 4; ++kk) {
      kstep(aj0, aj1, 2 * kk);
      kstep(aj1, aj0, 2 * kk + 1);
    }
    kstep(aj0, aj1, 8);

    const float inv9 = 1.0f / 9.0f;
#pragma unroll
    for (int m = 0; m < 6; ++m) {
      float* op = out + ((size_t)b * OC_ + g * 96 + 16 * m + r0) * N_ + col;
#pragma unroll
      for (int r = 0; r < 8; ++r) op[(size_t)r * N_] = acc[m][r] * inv9;
    }
  }
}

extern "C" void kernel_launch(void* const* d_in, const int* in_sizes, int n_in,
                              void* d_out, int out_size, void* d_ws, size_t ws_size,
                              hipStream_t stream) {
  (void)in_sizes; (void)n_in; (void)out_size; (void)ws_size;
  const float* x    = (const float*)d_in[0];  // [8,192,56,56] f32
  const float* W    = (const float*)d_in[1];  // [384,96] f32
  const float* bias = (const float*)d_in[2];  // [384] f32
  float* out = (float*)d_out;                 // [8,384,56,56] f32

  // Workspace layout (~20.3 MB total)
  char* ws = (char*)d_ws;
  const size_t XN_BYTES = (size_t)B_ * N_ * C_ * sizeof(bf16);     // 9,633,792
  const size_t SQ_BYTES = (size_t)B_ * N_ * sizeof(float);         //   100,352
  bf16*  xn = (bf16*)ws;
  bf16*  xt = (bf16*)(ws + XN_BYTES);
  float* sq = (float*)(ws + 2 * XN_BYTES);
  int*   nn = (int*)(ws + 2 * XN_BYTES + SQ_BYTES);                //   903,168

  prep_kernel<<<(B_ * N_ + 255) / 256, 256, 0, stream>>>(x, xn, xt, sq);
  knn_kernel<<<dim3(NT_, B_), 32, 0, stream>>>(xn, sq, nn);
  conv_kernel<<<dim3(N_ / 64, B_, 4), 128, 0, stream>>>(xt, W, bias, nn, out);
}
